// CrossAttention_44916767982026
// MI455X (gfx1250) — matrix operand order repaired and checked by reading, not compiled, not run on hardware
//
#include <hip/hip_runtime.h>

// ---------------------------------------------------------------------------
// CDNA5 (gfx1250) fused cross-attention with SiLU "softmax".
//   B=2, Sq=Sk=2048, HIDDEN=1024, HEADS=16, HEAD_DIM=64, SCALE=0.125
// Matmuls: v_wmma_f32_16x16x32_f16. GEMM tiles staged in LDS by the
// Tensor Data Mover (tensor_load_to_lds + s_wait_tensorcnt).
// ---------------------------------------------------------------------------

typedef __attribute__((ext_vector_type(8)))  _Float16 v8h;
typedef __attribute__((ext_vector_type(16))) _Float16 v16h;
typedef __attribute__((ext_vector_type(8)))  float    v8f;
typedef __attribute__((ext_vector_type(4)))  unsigned int u32x4;
typedef __attribute__((ext_vector_type(8)))  int      i32x8;
typedef __attribute__((ext_vector_type(4)))  int      i32x4;

#define HID   1024
#define SEQ   2048
#define NB    2
#define NH    16
#define HD    64
#define MTOT  (NB * SEQ)          // 4096 rows for [B*S, HID] matrices
#define QSCALE 0.125f             // HEAD_DIM^-0.5, folded into Q projection
#define NEGINF (-1.0e9f)

static __device__ __forceinline__ v16h cat8(v8h a, v8h b) {
  return __builtin_shufflevector(a, b, 0,1,2,3,4,5,6,7,8,9,10,11,12,13,14,15);
}

static __device__ __forceinline__ v8f wmma_f16(v16h a, v16h b, v8f c) {
  return __builtin_amdgcn_wmma_f32_16x16x32_f16(
      /*neg_a=*/false, a, /*neg_b=*/false, b,
      /*c_mod=*/(short)0, c, /*reuse_a=*/false, /*reuse_b=*/false);
}

// ---------------------------------------------------------------------------
// Tensor Data Mover: 2-D tile (rows x kElems of f16) global -> LDS.
// Descriptor bit-packing per CDNA5 ISA 8.3/8.4 (D# group0/group1).
// LDS result layout: packed [rows][kElems] row-major.
// ---------------------------------------------------------------------------
static __device__ __forceinline__ void tdm_load_2d_f16(
    unsigned lds_off, const _Float16* gptr,
    unsigned kElems, unsigned rows,
    unsigned tensor_w, unsigned tensor_h, unsigned row_stride_elems) {
  unsigned long long ga = (unsigned long long)(size_t)gptr;
  u32x4 g0;
  g0[0] = 1u;                                   // count=1 (valid user D#)
  g0[1] = lds_off;                              // LDS byte address
  g0[2] = (unsigned)ga;                         // global addr [31:0]
  g0[3] = (unsigned)(ga >> 32) | (2u << 30);    // addr [56:32] | type=2
  i32x8 g1;
  g1[0] = (int)(1u << 16);                      // data_size=1 -> 2 bytes
  g1[1] = (int)((tensor_w & 0xFFFFu) << 16);            // tensor_dim0 lo16
  g1[2] = (int)(((tensor_w >> 16) & 0xFFFFu) |
                ((tensor_h & 0xFFFFu) << 16));          // dim0 hi | dim1 lo
  g1[3] = (int)(((tensor_h >> 16) & 0xFFFFu) |
                ((kElems & 0xFFFFu) << 16));            // dim1 hi | tile_dim0
  g1[4] = (int)(rows & 0xFFFFu);                        // tile_dim1 (dim2=0)
  g1[5] = (int)row_stride_elems;                        // dim0_stride lo32
  g1[6] = 0;                                            // stride hi bits
  g1[7] = 0;
  i32x4 z4 = {0, 0, 0, 0};
#if __clang_major__ >= 23
  i32x8 z8 = {0, 0, 0, 0, 0, 0, 0, 0};
  __builtin_amdgcn_tensor_load_to_lds(g0, g1, z4, z4, z8, 0);
#else
  __builtin_amdgcn_tensor_load_to_lds(g0, g1, z4, z4, 0);
#endif
}

// ---------------------------------------------------------------------------
// fp32 -> fp16 elementwise conversion
// ---------------------------------------------------------------------------
__global__ void cvt_f32_f16(const float* __restrict__ x,
                            _Float16* __restrict__ y, int n) {
  int i = blockIdx.x * blockDim.x + threadIdx.x;
  if (i < n) y[i] = (_Float16)x[i];
}

// ---------------------------------------------------------------------------
// GEMM: Y[m,n] = (sum_k A[m,k] * W[n,k] + bias[n]) * oscale
//   A: f16 [4096,1024] rm ; W: f16 [1024,1024] rm ([out,in])
//   MODE 0: f16 out [4096,1024]
//   MODE 1: f16 out, V head-transposed: Vt[((b*16+h)*64+d)*2048 + s]
//   MODE 2: f32 out [4096,1024], no bias
// Block 256 (8 waves); wave w -> 32 rows; WG tile 256x64.
// A (256x32) and W (64x32) K-slabs DMA'd to LDS by TDM (wave 0).
// Grid: (4096/256, 1024/64) = (16, 16)
// ---------------------------------------------------------------------------
template <int MODE>
__global__ void gemm_wmma(const _Float16* __restrict__ A,
                          const _Float16* __restrict__ W,
                          const float* __restrict__ bias, float oscale,
                          _Float16* __restrict__ outH,
                          float* __restrict__ outF) {
  __shared__ _Float16 sA[256 * 32];   // 16 KB
  __shared__ _Float16 sB[64 * 32];    //  4 KB
  const int w    = threadIdx.x >> 5;
  const int lane = threadIdx.x & 31;
  const int r    = lane & 15;
  const int hi   = lane >> 4;

  const int m0 = blockIdx.x * 256;
  const int n0 = blockIdx.y * 64;

  const unsigned ldsA = (unsigned)(size_t)(&sA[0]);
  const unsigned ldsB = (unsigned)(size_t)(&sB[0]);

  v8f acc[2][4] = {};

  for (int kc = 0; kc < HID; kc += 32) {
    if (w == 0) {
      tdm_load_2d_f16(ldsA, A + ((size_t)m0 * HID + kc), 32, 256,
                      HID, MTOT, HID);
      tdm_load_2d_f16(ldsB, W + ((size_t)n0 * HID + kc), 32, 64,
                      HID, HID, HID);
      __builtin_amdgcn_s_wait_tensorcnt(0);
    }
    __syncthreads();   // tiles ready in LDS

    v16h af[2];
#pragma unroll
    for (int s = 0; s < 2; ++s) {
      // A-layout: lane holds row m = r, K = hi*16 .. +15 contiguous
      const _Float16* ap = sA + (size_t)(w * 32 + s * 16 + r) * 32 + hi * 16;
      af[s] = *(const v16h*)ap;
    }
#pragma unroll
    for (int nt = 0; nt < 4; ++nt) {
      // B-layout: lane holds column n = r, K = hi*16 .. +15 contiguous
      const _Float16* bp = sB + (size_t)(nt * 16 + r) * 32 + hi * 16;
      v16h bf = *(const v16h*)bp;
      acc[0][nt] = wmma_f16(af[0], bf, acc[0][nt]);
      acc[1][nt] = wmma_f16(af[1], bf, acc[1][nt]);
    }
    __syncthreads();   // all reads done before next DMA overwrites
  }

#pragma unroll
  for (int s = 0; s < 2; ++s) {
#pragma unroll
    for (int nt = 0; nt < 4; ++nt) {
      const int n = n0 + nt * 16 + r;
      const float bb = (MODE == 2) ? 0.0f : (bias ? bias[n] : 0.0f);
#pragma unroll
      for (int j = 0; j < 8; ++j) {
        const int m = m0 + w * 32 + s * 16 + j + hi * 8;
        const float v = (acc[s][nt][j] + bb) * oscale;
        if (MODE == 2) {
          outF[(size_t)m * HID + n] = v;
        } else if (MODE == 1) {
          const int b2 = m >> 11, s2 = m & (SEQ - 1);
          const int h2 = n >> 6,  d2 = n & (HD - 1);
          outH[(((size_t)(b2 * NH + h2) * HD + d2) << 11) + s2] = (_Float16)v;
        } else {
          outH[(size_t)m * HID + n] = (_Float16)v;
        }
      }
    }
  }
}

// ---------------------------------------------------------------------------
// Fused attention (scores pre-scaled via Q): per (b,h), 16 q rows per wave.
//   s = Q Kt ; mask==0 -> -1e9 ; silu ; out = attn @ V
// Grid: (2048/128, 32) ; block 256 (8 waves). LDS: 16x32 f16 per wave.
// ---------------------------------------------------------------------------
__global__ void attn_silu_wmma(const _Float16* __restrict__ Qh,
                               const _Float16* __restrict__ Kh,
                               const _Float16* __restrict__ Vt,
                               const int* __restrict__ mask,
                               _Float16* __restrict__ attn_h) {
  __shared__ _Float16 sbuf[8 * 16 * 32];   // 8 KB, wave-private tiles

  const int w    = threadIdx.x >> 5;
  const int lane = threadIdx.x & 31;
  const int r    = lane & 15;
  const int hi   = lane >> 4;

  const int b = blockIdx.y >> 4;
  const int h = blockIdx.y & (NH - 1);
  const int qbase = blockIdx.x * 128 + w * 16;

  _Float16* sl = sbuf + w * (16 * 32);

  // Q fragments for d = 0..31 and 32..63 (kept in registers, pre-scaled)
  const _Float16* qp =
      Qh + ((size_t)(b * SEQ + qbase + r)) * HID + h * HD + hi * 16;
  v16h aq0 = *(const v16h*)qp;
  v16h aq1 = *(const v16h*)(qp + 32);

  v8f acc[4] = {};
  const size_t maskBase = ((size_t)b * SEQ + qbase) * SEQ;
  const _Float16* vbase =
      Vt + ((size_t)(b * NH + h) * HD + r) * SEQ;   // row d = r (+ nt*16)

  for (int kb = 0; kb < SEQ; kb += 32) {
    // ---- issue all global loads for this 32-key block up front ----
    const _Float16* vp = vbase + kb + hi * 16;
    v16h vf0 = *(const v16h*)(vp);
    v16h vf1 = *(const v16h*)(vp + 16 * SEQ);
    v16h vf2 = *(const v16h*)(vp + 32 * SEQ);
    v16h vf3 = *(const v16h*)(vp + 48 * SEQ);

    const _Float16* kp0 =
        Kh + ((size_t)(b * SEQ + kb + r)) * HID + h * HD + hi * 16;
    const _Float16* kp1 = kp0 + (size_t)16 * HID;
    v16h ka0 = *(const v16h*)kp0;          // keys kb..kb+15,  d lo
    v16h ka1 = *(const v16h*)(kp0 + 32);   //                  d hi
    v16h kc0 = *(const v16h*)kp1;          // keys kb+16..+31, d lo
    v16h kc1 = *(const v16h*)(kp1 + 32);   //                  d hi

    int mk0[8], mk1[8];
#pragma unroll
    for (int j = 0; j < 8; ++j) {
      const size_t mrow = maskBase + (size_t)(j + hi * 8) * SEQ + kb + r;
      mk0[j] = mask[mrow];
      mk1[j] = mask[mrow + 16];
    }

    // ---- scores: two 16x16 tiles ----
    v8f s0 = {};
    s0 = wmma_f16(aq0, ka0, s0);
    s0 = wmma_f16(aq1, ka1, s0);
    v8f s1 = {};
    s1 = wmma_f16(aq0, kc0, s1);
    s1 = wmma_f16(aq1, kc1, s1);

    // ---- mask + silu, stash as row-major [q16][key32] tile in LDS ----
#pragma unroll
    for (int j = 0; j < 8; ++j) {
      const int m = j + hi * 8;
      float v0 = (mk0[j] == 0) ? NEGINF : s0[j];
      float v1 = (mk1[j] == 0) ? NEGINF : s1[j];
      const float a0 = v0 * __builtin_amdgcn_rcpf(1.0f + __expf(-v0));
      const float a1 = v1 * __builtin_amdgcn_rcpf(1.0f + __expf(-v1));
      sl[m * 32 + r]      = (_Float16)a0;
      sl[m * 32 + 16 + r] = (_Float16)a1;
    }

    // ---- reload scores in A-fragment layout (wave-private, DS in-order) ----
    const _Float16* sp = sl + r * 32 + hi * 16;
    v16h as = *(const v16h*)sp;

    // ---- out += attn @ V ----
    acc[0] = wmma_f16(as, vf0, acc[0]);
    acc[1] = wmma_f16(as, vf1, acc[1]);
    acc[2] = wmma_f16(as, vf2, acc[2]);
    acc[3] = wmma_f16(as, vf3, acc[3]);
  }

  // Store 16x64 f32 accumulators as f16 into [4096, 1024] layout
#pragma unroll
  for (int nt = 0; nt < 4; ++nt) {
#pragma unroll
    for (int j = 0; j < 8; ++j) {
      const int q = qbase + j + hi * 8;
      const int c = h * HD + nt * 16 + r;
      attn_h[(size_t)(b * SEQ + q) * HID + c] = (_Float16)acc[nt][j];
    }
  }
}

// ---------------------------------------------------------------------------
// Host-side orchestration
// ---------------------------------------------------------------------------
static inline void launch_cvt(const float* x, _Float16* y, int n,
                              hipStream_t stream) {
  cvt_f32_f16<<<(n + 255) / 256, 256, 0, stream>>>(x, y, n);
}

extern "C" void kernel_launch(void* const* d_in, const int* in_sizes, int n_in,
                              void* d_out, int out_size, void* d_ws,
                              size_t ws_size, hipStream_t stream) {
  const float* query = (const float*)d_in[0];
  const float* key   = (const float*)d_in[1];
  const float* value = (const float*)d_in[2];
  const int*   mask  = (const int*)d_in[3];
  const float* Wq    = (const float*)d_in[4];
  const float* bq    = (const float*)d_in[5];
  const float* Wk    = (const float*)d_in[6];
  const float* bk    = (const float*)d_in[7];
  const float* Wv    = (const float*)d_in[8];
  const float* bv    = (const float*)d_in[9];
  const float* Wo    = (const float*)d_in[10];
  float* out = (float*)d_out;

  const size_t MB = 1024 * 1024;
  char* ws = (char*)d_ws;
  _Float16* q_h    = (_Float16*)(ws + 0 * MB);   // 8 MB each
  _Float16* k_h    = (_Float16*)(ws + 8 * MB);
  _Float16* v_h    = (_Float16*)(ws + 16 * MB);
  _Float16* Qh     = (_Float16*)(ws + 24 * MB);
  _Float16* Kh     = (_Float16*)(ws + 32 * MB);
  _Float16* Vt     = (_Float16*)(ws + 40 * MB);
  _Float16* attn_h = (_Float16*)(ws + 48 * MB);
  _Float16* Wq_h   = (_Float16*)(ws + 56 * MB);  // 2 MB each
  _Float16* Wk_h   = (_Float16*)(ws + 58 * MB);
  _Float16* Wv_h   = (_Float16*)(ws + 60 * MB);
  _Float16* Wo_h   = (_Float16*)(ws + 62 * MB);

  const int nAct = NB * SEQ * HID;   // 4,194,304
  const int nW   = HID * HID;        // 1,048,576

  launch_cvt(query, q_h, nAct, stream);
  launch_cvt(key,   k_h, nAct, stream);
  launch_cvt(value, v_h, nAct, stream);
  launch_cvt(Wq, Wq_h, nW, stream);
  launch_cvt(Wk, Wk_h, nW, stream);
  launch_cvt(Wv, Wv_h, nW, stream);
  launch_cvt(Wo, Wo_h, nW, stream);

  dim3 gGrid(MTOT / 256, HID / 64);   // (16, 16)
  dim3 blk(256);
  // SCALE folded into Q projection epilogue
  gemm_wmma<0><<<gGrid, blk, 0, stream>>>(q_h, Wq_h, bq, QSCALE, Qh, nullptr);
  gemm_wmma<0><<<gGrid, blk, 0, stream>>>(k_h, Wk_h, bk, 1.0f,   Kh, nullptr);
  gemm_wmma<1><<<gGrid, blk, 0, stream>>>(v_h, Wv_h, bv, 1.0f,   Vt, nullptr);

  dim3 aGrid(SEQ / 128, NB * NH);     // (16, 32)
  attn_silu_wmma<<<aGrid, blk, 0, stream>>>(Qh, Kh, Vt, mask, attn_h);

  gemm_wmma<2><<<gGrid, blk, 0, stream>>>(attn_h, Wo_h, nullptr, 1.0f,
                                          nullptr, out);
}
